// NN_45535243272834
// MI455X (gfx1250) — matrix operand order repaired and checked by reading, not compiled, not run on hardware
//
#include <hip/hip_runtime.h>

#define NPTS    1000000
#define NTILES  (NPTS / 16)    // 62500
#define NGROUPS (NTILES / 2)   // 31250, exact (2 tiles per iteration)
#define NFRAG   46             // W1:4  W2..W6:40  W7:2

typedef __attribute__((ext_vector_type(16))) _Float16 v16h;
typedef __attribute__((ext_vector_type(8)))  float    v8f;
typedef __attribute__((ext_vector_type(8)))  unsigned v8u;

__device__ __forceinline__ unsigned pkh(float a, float b) {
    // v_cvt_pk_rtz_f16_f32: low 16 bits = a, high = b
    return __builtin_bit_cast(unsigned, __builtin_amdgcn_cvt_pkrtz(a, b));
}
__device__ __forceinline__ unsigned swap16(unsigned v) {
    // ds_swizzle group-of-32: and=0x1f, or=0, xor=0x10 -> lane ^= 16
    return (unsigned)__builtin_amdgcn_ds_swizzle((int)v, 0x401F);
}
__device__ __forceinline__ float silu(float s) {
    float e = __builtin_amdgcn_exp2f(-1.44269504088896340736f * s);
    return s * __builtin_amdgcn_rcpf(1.0f + e);
}
__device__ __forceinline__ v8f wmma(v16h a, v16h b, v8f c) {
    return __builtin_amdgcn_wmma_f32_16x16x32_f16(false, a, false, b, (short)0, c,
                                                  false, false);
}

// Steady-state fragment fetch: one aligned 32B LDS read per lane (2x ds_load_b128).
__device__ __forceinline__ v16h ld_frag(const unsigned* sF, int fid, int lane) {
    v8u u = *(const v8u*)(sF + fid * 256 + lane * 8);
    return __builtin_bit_cast(v16h, u);
}

// SiLU-activate two raw D tiles and pack into a B fragment (32x16 f16, K x N).
// B layout: VGPR j: lanes 0-15 hold K=2j,2j+1 ; lanes 16-31 hold K=16+2j,+1.
// D layout: lanes 0-15 VGPR r = feat r ; lanes 16-31 VGPR r = feat r+8.
__device__ __forceinline__ v16h make_b_act(const v8f& Da, const v8f& Db, bool lo) {
    v8u u;
#pragma unroll
    for (int i = 0; i < 4; ++i) {
        unsigned pa = pkh(silu(Da[2 * i]), silu(Da[2 * i + 1]));
        unsigned pb = pkh(silu(Db[2 * i]), silu(Db[2 * i + 1]));
        unsigned sa = swap16(pa);
        unsigned sb = swap16(pb);
        u[i]     = lo ? pa : sb;   // K=2i   (lo) | K=16+2i (hi)
        u[i + 4] = lo ? sa : pb;   // K=8+2i (lo) | K=24+2i (hi)
    }
    return __builtin_bit_cast(v16h, u);
}

// Bias as initial C accumulator: C VGPR r = b[t*16 + r + (hi ? 8 : 0)]
__device__ __forceinline__ v8f load_c(const float* sBl, int t, bool lo) {
    const float* p = sBl + t * 16 + (lo ? 0 : 8);
    v8f c;
#pragma unroll
    for (int r = 0; r < 8; ++r) c[r] = p[r];
    return c;
}

__global__ void __launch_bounds__(256, 4)
mlp60_wmma_kernel(const float* __restrict__ x,  const float* __restrict__ y,
                  const float* __restrict__ W1, const float* __restrict__ b1,
                  const float* __restrict__ W2, const float* __restrict__ b2,
                  const float* __restrict__ W3, const float* __restrict__ b3,
                  const float* __restrict__ W4, const float* __restrict__ b4,
                  const float* __restrict__ W5, const float* __restrict__ b5,
                  const float* __restrict__ W6, const float* __restrict__ b6,
                  const float* __restrict__ W7, const float* __restrict__ b7,
                  float* __restrict__ out) {
    // Pre-packed weight fragments: [frag][lane][8 dwords] (32B per lane slot)
    __shared__ __align__(32) unsigned sFrag[NFRAG * 256];   // 47104 B
    __shared__ __align__(16) float    sB[6 * 64];           // biases, zero padded
    __shared__ float                  sB7[1];

    const int  tid  = threadIdx.x;
    const int  lane = tid & 31;
    const bool lo   = (lane & 16) == 0;

    // ---- one-pass prepack: biases + per-lane A-fragment bytes ----
    const float* Bs[6] = {b1, b2, b3, b4, b5, b6};
#pragma unroll
    for (int l = 0; l < 6; ++l)
        if (tid < 64) sB[l * 64 + tid] = (tid < 60) ? Bs[l][tid] : 0.f;
    if (tid == 0) sB7[0] = b7[0];

    // frag ids: 0..3 = W1(t,k=0) ; 4..43 = W2..W6 (4 + l*8 + k*4 + t) ; 44..45 = W7(k)
    for (int s = tid; s < NFRAG * 32; s += 256) {
        const int fid = s >> 5;
        const int ln  = s & 31;
        const float* W; int rows, cols, t, kt;
        if (fid < 4)        { W = W1; rows = 60; cols = 2;  t = fid;   kt = 0; }
        else if (fid < 44)  {
            int f = fid - 4, li = f >> 3;
            W = (li == 0) ? W2 : (li == 1) ? W3 : (li == 2) ? W4 : (li == 3) ? W5 : W6;
            rows = 60; cols = 60; kt = (f >> 2) & 1; t = f & 3;
        }
        else                { W = W7; rows = 1;  cols = 60; t = 0;     kt = fid - 44; }

        const int o  = t * 16 + (ln & 15);
        const int h8 = (ln & 16) ? 16 : 0;
#pragma unroll
        for (int v = 0; v < 8; ++v) {
            int kk = 2 * v + h8 + kt * 32;
            float a = 0.f, b = 0.f;
            if (o < rows && kk < cols) {   // kk even -> kk+1 also < cols
                a = W[o * cols + kk];
                b = W[o * cols + kk + 1];
            }
            sFrag[fid * 256 + ln * 8 + v] = pkh(a, b);
        }
    }
    __syncthreads();

    const float b7v = sB7[0];

    // ---- grid-stride loop over groups of 2 point-tiles (32 points) ----
    const int wave = blockIdx.x * (blockDim.x >> 5) + (tid >> 5);
    const int nw   = gridDim.x * (blockDim.x >> 5);

    for (int g = wave; g < NGROUPS; g += nw) {
        // Compiler-level memory barrier: stops LICM from hoisting the (loop-
        // invariant) LDS fragment/bias loads out of the loop, which would
        // otherwise recreate a >256-VGPR live set and spill to scratch.
        asm volatile("" ::: "memory");

        const int p0 = g * 32;
        const int pn = p0 + nw * 32;
        if (pn < NPTS) {                       // wave-uniform
            __builtin_prefetch(&x[pn], 0, 0);  // global_prefetch_b8
            __builtin_prefetch(&y[pn], 0, 0);
        }

        // Layer 1 inputs: tile0 = points p0.., tile1 = points p0+16..
        float xa = x[p0 + (lane & 15)],      ya = y[p0 + (lane & 15)];
        float xb = x[p0 + 16 + (lane & 15)], yb = y[p0 + 16 + (lane & 15)];
        v8u ba = {0u,0u,0u,0u,0u,0u,0u,0u}, bb = ba;
        ba[0] = lo ? pkh(xa, ya) : 0u;
        bb[0] = lo ? pkh(xb, yb) : 0u;
        v16h B0a = __builtin_bit_cast(v16h, ba);
        v16h B0b = __builtin_bit_cast(v16h, bb);

        v8f D0[4], D1[4];
#pragma unroll
        for (int t = 0; t < 4; ++t) {
            v8f  c = load_c(sB, t, lo);
            v16h a = ld_frag(sFrag, t, lane);
            D0[t] = wmma(a, B0a, c);
            D1[t] = wmma(a, B0b, c);
        }

        // Layers 2..6
#pragma unroll
        for (int l = 0; l < 5; ++l) {
            // tile0 activation+pack first, then tile1 (keeps peak pressure low)
            v16h Blo0 = make_b_act(D0[0], D0[1], lo), Bhi0 = make_b_act(D0[2], D0[3], lo);
            v16h Blo1 = make_b_act(D1[0], D1[1], lo), Bhi1 = make_b_act(D1[2], D1[3], lo);
#pragma unroll
            for (int t = 0; t < 4; ++t) {
                v8f  c  = load_c(sB + (l + 1) * 64, t, lo);
                v16h a0 = ld_frag(sFrag, 4 + l * 8 + t,     lane);
                v16h a1 = ld_frag(sFrag, 4 + l * 8 + 4 + t, lane);
                D0[t] = wmma(a1, Bhi0, wmma(a0, Blo0, c));
                D1[t] = wmma(a1, Bhi1, wmma(a0, Blo1, c));
            }
        }

        // Layer 7: out = h . W7 + b7  (only M=0 row of A non-zero)
        v16h Blo0 = make_b_act(D0[0], D0[1], lo), Bhi0 = make_b_act(D0[2], D0[3], lo);
        v16h Blo1 = make_b_act(D1[0], D1[1], lo), Bhi1 = make_b_act(D1[2], D1[3], lo);
        v16h a70 = ld_frag(sFrag, 44, lane), a71 = ld_frag(sFrag, 45, lane);
        v8f c7a = {}, c7b = {};
        c7a[0] = lo ? b7v : 0.f;
        c7b[0] = c7a[0];
        c7a = wmma(a71, Bhi0, wmma(a70, Blo0, c7a));
        c7b = wmma(a71, Bhi1, wmma(a70, Blo1, c7b));

        // Full-wave coalesced store: lanes 0-15 -> tile0, lanes 16-31 -> tile1
        unsigned r0 = __builtin_bit_cast(unsigned, c7a[0]);
        unsigned r1 = swap16(__builtin_bit_cast(unsigned, c7b[0]));
        unsigned rv = lo ? r0 : r1;
        out[p0 + lane] = __builtin_bit_cast(float, rv);
    }
}

extern "C" void kernel_launch(void* const* d_in, const int* in_sizes, int n_in,
                              void* d_out, int out_size, void* d_ws, size_t ws_size,
                              hipStream_t stream) {
    (void)in_sizes; (void)n_in; (void)d_ws; (void)ws_size; (void)out_size;
    const float* x  = (const float*)d_in[0];
    const float* y  = (const float*)d_in[1];
    const float* W1 = (const float*)d_in[2];
    const float* b1 = (const float*)d_in[3];
    const float* W2 = (const float*)d_in[4];
    const float* b2 = (const float*)d_in[5];
    const float* W3 = (const float*)d_in[6];
    const float* b3 = (const float*)d_in[7];
    const float* W4 = (const float*)d_in[8];
    const float* b4 = (const float*)d_in[9];
    const float* W5 = (const float*)d_in[10];
    const float* b5 = (const float*)d_in[11];
    const float* W6 = (const float*)d_in[12];
    const float* b6 = (const float*)d_in[13];
    const float* W7 = (const float*)d_in[14];
    const float* b7 = (const float*)d_in[15];

    dim3 grid(512), block(256);   // 4096 waves, ~7.6 groups/wave
    mlp60_wmma_kernel<<<grid, block, 0, stream>>>(
        x, y, W1, b1, W2, b2, W3, b3, W4, b4, W5, b5, W6, b6, W7, b7,
        (float*)d_out);
}
